// BootstrapRecurrentMixer_36867999269440
// MI455X (gfx1250) — compile-verified, matrix operand order checked by reading, not run on hardware
//
#include <hip/hip_runtime.h>

// ---------------------------------------------------------------- constants
#define MODEL_DIM 1024
#define INNER     2048
#define STATE_DIM 256
#define BATCH     4
#define SEQ       2048
#define M_TOK     (BATCH * SEQ)      // 8192 tokens
#define N_IN      (3 * INNER)        // 6144
#define NWG       32                 // scan workgroups (persistent)
#define IW        (INNER / NWG)      // 64 inner cols per scan WG
#define LDS_STRIDE 40                // halves per LDS tile row (bank-spread, 16B aligned)

typedef _Float16 h8  __attribute__((ext_vector_type(8)));
typedef _Float16 h16 __attribute__((ext_vector_type(16)));
typedef float    f8  __attribute__((ext_vector_type(8)));

// Branch-free tanh/sigmoid on v_exp_f32 (overflow-safe, ~1e-6 rel err)
__device__ __forceinline__ float fast_tanh(float x) {
  x = fminf(fmaxf(x, -15.f), 15.f);
  float e = __expf(2.f * x);
  return (e - 1.f) / (e + 1.f);
}
__device__ __forceinline__ float fast_sigmoid(float x) {
  return 1.f / (1.f + __expf(-x));
}

// ------------------------------------------------------------ LDS fragment
// 16-bit fragment (16 rows x 32 k), LDS tile row stride = LDS_STRIDE halves:
//   lanes 0-15  : row = r+lane,    halves 0-7 = k[0..7],  8-15 = k[16..23]
//   lanes 16-31 : row = r+lane-16, halves 0-7 = k[8..15], 8-15 = k[24..31]
__device__ __forceinline__ h16 frag_lds(const _Float16* buf, int r, int lane) {
  int row  = r + (lane & 15);
  int koff = (lane >> 4) << 3;                 // 0 or 8
  const _Float16* p = buf + row * LDS_STRIDE + koff;
  h8 lo = *(const h8*)p;
  h8 hi = *(const h8*)(p + 16);
  return __builtin_shufflevector(lo, hi, 0,1,2,3,4,5,6,7,8,9,10,11,12,13,14,15);
}

// ------------------------------------------------------------ WMMA GEMM
// C[M,N] = A[M,K](f16) * BT[N,K](f16)^T + bias
// Block 256 threads (8 waves), block tile 128x128, wave tile 32x64, BK=32.
// Double-buffered LDS staging: global loads for step k+1 issued before the
// WMMAs of step k; ds_store after compute; global latency hidden per block.
// MODE 1: in_proj epilogue -> split cols into U | G | tanh(V), f32, ld=INNER
// MODE 2: out_proj epilogue -> plain f32 store, ld = N
template <int MODE>
__global__ __launch_bounds__(256) void gemm_wmma(
    const _Float16* __restrict__ A, const _Float16* __restrict__ BT,
    const float* __restrict__ bias,
    float* __restrict__ out0, float* __restrict__ out1, float* __restrict__ out2,
    int M, int N, int K) {
  __shared__ _Float16 Ab[2][128 * LDS_STRIDE];   // 10 KB each
  __shared__ _Float16 Bb[2][128 * LDS_STRIDE];

  const int tid  = threadIdx.x;
  const int lane = tid & 31;
  const int wave = tid >> 5;
  const int m_blk = blockIdx.y * 128;
  const int n_blk = blockIdx.x * 128;
  const int wm = (wave >> 1) * 32;               // wave M offset in tile
  const int wn = (wave & 1) * 64;                // wave N offset in tile

  // cooperative tile-load mapping: thread -> (row pair, 16B chunk)
  const int lrow = tid >> 2;                     // 0..63
  const int c8   = (tid & 3) * 8;                // half offset 0,8,16,24
  const _Float16* gA0 = A  + (size_t)(m_blk + lrow)      * K + c8;
  const _Float16* gA1 = A  + (size_t)(m_blk + lrow + 64) * K + c8;
  const _Float16* gB0 = BT + (size_t)(n_blk + lrow)      * K + c8;
  const _Float16* gB1 = BT + (size_t)(n_blk + lrow + 64) * K + c8;
  _Float16* lA0 = &Ab[0][lrow * LDS_STRIDE + c8];
  _Float16* lA1 = &Ab[0][(lrow + 64) * LDS_STRIDE + c8];
  _Float16* lB0 = &Bb[0][lrow * LDS_STRIDE + c8];
  _Float16* lB1 = &Bb[0][(lrow + 64) * LDS_STRIDE + c8];
  const int lds_buf = 128 * LDS_STRIDE;          // halves per buffer

  f8 acc[2][4];
  #pragma unroll
  for (int i = 0; i < 2; ++i)
    #pragma unroll
    for (int j = 0; j < 4; ++j) acc[i][j] = (f8)0.f;

  // prologue: stage k=0 tile into buffer 0
  {
    h8 ra0 = *(const h8*)gA0, ra1 = *(const h8*)gA1;
    h8 rb0 = *(const h8*)gB0, rb1 = *(const h8*)gB1;
    *(h8*)lA0 = ra0; *(h8*)lA1 = ra1;
    *(h8*)lB0 = rb0; *(h8*)lB1 = rb1;
  }
  __syncthreads();

  const int NK = K >> 5;
  for (int kt = 0; kt < NK; ++kt) {
    const int cur = kt & 1;
    const int nxt = cur ^ 1;

    // issue next tile's global loads early (consumed after the WMMAs)
    h8 ra0, ra1, rb0, rb1;
    if (kt + 1 < NK) {
      int ko = (kt + 1) << 5;
      ra0 = *(const h8*)(gA0 + ko); ra1 = *(const h8*)(gA1 + ko);
      rb0 = *(const h8*)(gB0 + ko); rb1 = *(const h8*)(gB1 + ko);
    }
    if (kt + 2 < NK) {                           // gfx1250 global_prefetch_b8
      int kp = (kt + 2) << 5;
      __builtin_prefetch(gA0 + kp, 0, 0);
      __builtin_prefetch(gB0 + kp, 0, 0);
    }

    // fragments from LDS
    h16 a0 = frag_lds(Ab[cur], wm,      lane);
    h16 a1 = frag_lds(Ab[cur], wm + 16, lane);
    h16 b[4];
    #pragma unroll
    for (int j = 0; j < 4; ++j) b[j] = frag_lds(Bb[cur], wn + 16 * j, lane);

    #pragma unroll
    for (int j = 0; j < 4; ++j) {
      acc[0][j] = __builtin_amdgcn_wmma_f32_16x16x32_f16(
          false, a0, false, b[j], (short)0, acc[0][j], false, false);
      acc[1][j] = __builtin_amdgcn_wmma_f32_16x16x32_f16(
          false, a1, false, b[j], (short)0, acc[1][j], false, false);
    }

    __syncthreads();                             // all waves done reading cur
    if (kt + 1 < NK) {
      *(h8*)(lA0 + nxt * lds_buf) = ra0; *(h8*)(lA1 + nxt * lds_buf) = ra1;
      *(h8*)(lB0 + nxt * lds_buf) = rb0; *(h8*)(lB1 + nxt * lds_buf) = rb1;
      __syncthreads();                           // next buffer visible
    }
  }

  // C layout: half g of lane holds (row = g + (lane<16?0:8), col = lane&15)
  const int m0 = m_blk + wm, n0 = n_blk + wn;
  #pragma unroll
  for (int mi = 0; mi < 2; ++mi) {
    #pragma unroll
    for (int ni = 0; ni < 4; ++ni) {
      int col = n0 + ni * 16 + (lane & 15);
      float bv = bias[col];
      #pragma unroll
      for (int g = 0; g < 8; ++g) {
        int row = m0 + mi * 16 + g + ((lane >> 4) << 3);
        float v = acc[mi][ni][g] + bv;
        if (MODE == 1) {
          int part = col >> 11;                  // 0:U 1:G 2:V
          int cc   = col & (INNER - 1);
          float* dst = (part == 0) ? out0 : (part == 1) ? out1 : out2;
          if (part == 2) v = fast_tanh(v);
          dst[(size_t)row * INNER + cc] = v;
        } else {
          out0[(size_t)row * N + col] = v;
        }
      }
    }
  }
}

// ------------------------------------------------------------ converts
__global__ void k_f32_to_f16(const float* __restrict__ src,
                             _Float16* __restrict__ dst, int n) {
  for (int i = blockIdx.x * blockDim.x + threadIdx.x; i < n;
       i += gridDim.x * blockDim.x)
    dst[i] = (_Float16)src[i];
}

// W[K][N] f32 -> WT[N][K] f16 (32x32 LDS tile, block 32x8)
__global__ void k_transpose_f16(const float* __restrict__ W,
                                _Float16* __restrict__ WT, int K, int N) {
  __shared__ float tile[32][33];
  int kb = blockIdx.y * 32, nb = blockIdx.x * 32;
  int tx = threadIdx.x, ty = threadIdx.y;
  #pragma unroll
  for (int i = 0; i < 32; i += 8)
    tile[ty + i][tx] = W[(size_t)(kb + ty + i) * N + nb + tx];
  __syncthreads();
  #pragma unroll
  for (int i = 0; i < 32; i += 8)
    WT[(size_t)(nb + ty + i) * K + kb + tx] = (_Float16)tile[tx][ty + i];
}

__global__ void k_zero_i32(int* __restrict__ p, int n) {
  for (int i = blockIdx.x * blockDim.x + threadIdx.x; i < n;
       i += gridDim.x * blockDim.x)
    p[i] = 0;
}

// ------------------------------------------------------------ scan kernel
// Persistent NWG=32 workgroups, 512 threads each. Weights LDS-resident.
// Cross-WG step barrier: double-buffered partial-state array + done[] counters
// with device-scope release/acquire.
__global__ __launch_bounds__(512, 1) void scan_kernel(
    const float* __restrict__ U, const float* __restrict__ G,
    const float* __restrict__ TV,
    const float* __restrict__ sp_w, const float* __restrict__ sp_b,
    const float* __restrict__ su_w, const float* __restrict__ su_b,
    _Float16* __restrict__ MIX16,
    float* __restrict__ partials,          // [2][NWG][BATCH][STATE_DIM]
    int* __restrict__ done,                // [SEQ]
    float* __restrict__ final_state) {     // [BATCH][STATE_DIM]
  __shared__ _Float16 s_spw[STATE_DIM * 2 * IW];   // [k][ug*IW+j]   64 KB
  __shared__ float    s_suw[IW * STATE_DIM];       // [j][n]         64 KB
  __shared__ float    s_state[BATCH * STATE_DIM];
  __shared__ float    s_sp[2 * BATCH * IW];
  __shared__ float    s_mix[BATCH * IW];

  const int wg    = blockIdx.x;
  const int tid   = threadIdx.x;
  const int jbase = wg * IW;

  for (int idx = tid; idx < STATE_DIM * IW; idx += 512) {
    int k = idx / IW, j = idx % IW;
    s_spw[k * (2 * IW) + j]      = (_Float16)sp_w[(size_t)k * (2 * INNER) + jbase + j];
    s_spw[k * (2 * IW) + IW + j] = (_Float16)sp_w[(size_t)k * (2 * INNER) + INNER + jbase + j];
  }
  for (int idx = tid; idx < IW * STATE_DIM; idx += 512) {
    int j = idx / STATE_DIM, n = idx % STATE_DIM;
    s_suw[j * STATE_DIM + n] = su_w[(size_t)(jbase + j) * STATE_DIM + n];
  }
  __syncthreads();

  for (int t = 0; t < SEQ; ++t) {
    // ---- 1) materialize state_t in LDS
    if (t == 0) {
      for (int i = tid; i < BATCH * STATE_DIM; i += 512) s_state[i] = 0.f;
    } else {
      if (tid == 0) {
        while (__hip_atomic_load(&done[t - 1], __ATOMIC_ACQUIRE,
                                 __HIP_MEMORY_SCOPE_AGENT) < NWG)
          __builtin_amdgcn_s_sleep(2);
      }
      __syncthreads();
      const float* pb = partials + (size_t)((t - 1) & 1) * NWG * BATCH * STATE_DIM;
      for (int i = tid; i < BATCH * STATE_DIM; i += 512) {
        float a = su_b[i & (STATE_DIM - 1)];
        #pragma unroll 4
        for (int w = 0; w < NWG; ++w) a += pb[w * BATCH * STATE_DIM + i];
        s_state[i] = a;
      }
    }
    __syncthreads();

    // ---- 2) sp = state @ sp_w slice + sp_b : one output per thread
    {
      int b  = tid >> 7;
      int ug = (tid >> 6) & 1;
      int j  = tid & (IW - 1);
      float a = sp_b[ug * INNER + jbase + j];
      const float* st = s_state + b * STATE_DIM;
      const _Float16* wp = s_spw + ug * IW + j;
      #pragma unroll 8
      for (int k = 0; k < STATE_DIM; ++k)
        a += st[k] * (float)wp[k * (2 * IW)];
      s_sp[(ug * BATCH + b) * IW + j] = a;
    }
    __syncthreads();

    // ---- 3) gate/candidate/mix for 256 (b,j) pairs
    if (tid < BATCH * IW) {
      int b = tid >> 6, j = tid & (IW - 1);
      size_t base = ((size_t)b * SEQ + t) * INNER + jbase + j;
      float u = U[base], g = G[base], tv = TV[base];
      float cand = fast_tanh(u + s_sp[(0 * BATCH + b) * IW + j]);
      float gate = fast_sigmoid(g + s_sp[(1 * BATCH + b) * IW + j]);
      float mv   = gate * cand + (1.f - gate) * tv;
      s_mix[tid]  = mv;
      MIX16[base] = (_Float16)mv;       // f16 operand for out_proj WMMA
    }
    __syncthreads();

    // ---- 4) partial new_state = mix @ su_w slice : 2 outputs per thread
    {
      float* pout = partials + (size_t)(t & 1) * NWG * BATCH * STATE_DIM
                  + (size_t)wg * BATCH * STATE_DIM;
      for (int i = tid; i < BATCH * STATE_DIM; i += 512) {
        int b = i >> 8, n = i & (STATE_DIM - 1);
        float a = 0.f;
        const float* mx = s_mix + b * IW;
        const float* wp = s_suw + n;
        #pragma unroll 8
        for (int j = 0; j < IW; ++j) a += mx[j] * wp[j * STATE_DIM];
        pout[i] = a;
      }
    }

    // ---- 5) publish (release)
    __threadfence();
    __syncthreads();
    if (tid == 0)
      __hip_atomic_fetch_add(&done[t], 1, __ATOMIC_RELEASE,
                             __HIP_MEMORY_SCOPE_AGENT);
  }

  // ---- final state
  if (wg == 0) {
    if (tid == 0) {
      while (__hip_atomic_load(&done[SEQ - 1], __ATOMIC_ACQUIRE,
                               __HIP_MEMORY_SCOPE_AGENT) < NWG)
        __builtin_amdgcn_s_sleep(2);
    }
    __syncthreads();
    const float* pb = partials + (size_t)((SEQ - 1) & 1) * NWG * BATCH * STATE_DIM;
    for (int i = tid; i < BATCH * STATE_DIM; i += 512) {
      float a = su_b[i & (STATE_DIM - 1)];
      for (int w = 0; w < NWG; ++w) a += pb[w * BATCH * STATE_DIM + i];
      final_state[i] = a;
    }
  }
}

// ------------------------------------------------------------ launcher
extern "C" void kernel_launch(void* const* d_in, const int* in_sizes, int n_in,
                              void* d_out, int out_size, void* d_ws,
                              size_t ws_size, hipStream_t stream) {
  (void)in_sizes; (void)n_in; (void)out_size; (void)ws_size;
  const float* hidden = (const float*)d_in[0];
  const float* in_w   = (const float*)d_in[1];
  const float* in_b   = (const float*)d_in[2];
  const float* sp_w   = (const float*)d_in[3];
  const float* sp_b   = (const float*)d_in[4];
  const float* su_w   = (const float*)d_in[5];
  const float* su_b   = (const float*)d_in[6];
  const float* out_w  = (const float*)d_in[7];
  const float* out_b  = (const float*)d_in[8];

  char* ws = (char*)d_ws;
  size_t off = 0;
  auto take = [&](size_t bytes) { char* p = ws + off; off += (bytes + 255) & ~(size_t)255; return p; };

  _Float16* A16   = (_Float16*)take((size_t)M_TOK * MODEL_DIM * 2);   // hidden f16
  _Float16* W1T   = (_Float16*)take((size_t)N_IN * MODEL_DIM * 2);    // in_w^T f16
  _Float16* WOT   = (_Float16*)take((size_t)MODEL_DIM * INNER * 2);   // out_w^T f16
  float*    Ubuf  = (float*)take((size_t)M_TOK * INNER * 4);
  float*    Gbuf  = (float*)take((size_t)M_TOK * INNER * 4);
  float*    TVbuf = (float*)take((size_t)M_TOK * INNER * 4);
  _Float16* MIX16 = (_Float16*)take((size_t)M_TOK * INNER * 2);
  float*    parts = (float*)take((size_t)2 * NWG * BATCH * STATE_DIM * 4);
  int*      done  = (int*)take((size_t)SEQ * 4);

  float* out_main  = (float*)d_out;                        // [M_TOK, MODEL_DIM]
  float* out_state = out_main + (size_t)M_TOK * MODEL_DIM; // [BATCH, STATE_DIM]

  // 1) precision converts / transposes
  k_f32_to_f16<<<2048, 256, 0, stream>>>(hidden, A16, M_TOK * MODEL_DIM);
  k_transpose_f16<<<dim3(N_IN / 32, MODEL_DIM / 32), dim3(32, 8), 0, stream>>>(
      in_w, W1T, MODEL_DIM, N_IN);
  k_transpose_f16<<<dim3(MODEL_DIM / 32, INNER / 32), dim3(32, 8), 0, stream>>>(
      out_w, WOT, INNER, MODEL_DIM);
  k_zero_i32<<<8, 256, 0, stream>>>(done, SEQ);

  // 2) in_proj GEMM (WMMA) + split/tanh epilogue
  gemm_wmma<1><<<dim3(N_IN / 128, M_TOK / 128), 256, 0, stream>>>(
      A16, W1T, in_b, Ubuf, Gbuf, TVbuf, M_TOK, N_IN, MODEL_DIM);

  // 3) sequential scan: persistent 32-WG kernel, LDS-resident weights
  scan_kernel<<<NWG, 512, 0, stream>>>(Ubuf, Gbuf, TVbuf, sp_w, sp_b, su_w,
                                       su_b, MIX16, parts, done, out_state);

  // 4) out_proj GEMM (WMMA)
  gemm_wmma<2><<<dim3(MODEL_DIM / 128, M_TOK / 128), 256, 0, stream>>>(
      MIX16, WOT, out_b, out_main, nullptr, nullptr, M_TOK, MODEL_DIM, INNER);
}
